// GPT1_fourier_60894046323346
// MI455X (gfx1250) — matrix operand order repaired and checked by reading, not compiled, not run on hardware
//
#include <hip/hip_runtime.h>
#include <cstdint>

// ---------------------------------------------------------------------------
// GPT1_fourier for MI455X (gfx1250).
// Transformer GEMMs: v_wmma_f32_16x16x32_bf16, register-blocked 2x2 tiles per
// wave (32x32 output / wave, 128x64 / 256-thread block), with A/B panels
// staged into LDS via global_load_async_to_lds_b128 (ASYNCcnt) and consumed
// with ds_load_b128.  Weights are converted fp32->bf16 transposed once per
// layer so all fragment traffic is contiguous 16B loads.
// ---------------------------------------------------------------------------

#define USE_ASYNC_LDS 1   // set 0 to fall back to global_load+ds_store staging

typedef __bf16 bf16_t;
typedef __attribute__((ext_vector_type(16))) __bf16 v16bf;
typedef __attribute__((ext_vector_type(8)))  float  v8f;

#define D_MODEL 512
#define SEQ_LEN 128
#define NLAYER  8

// float -> bf16, round-to-nearest-even
__device__ __forceinline__ bf16_t f2bf(float f) {
  unsigned u = __float_as_uint(f);
  unsigned r = (u + 0x7FFFu + ((u >> 16) & 1u)) >> 16;
  return __builtin_bit_cast(bf16_t, (unsigned short)r);
}

#if USE_ASYNC_LDS
// One 16-byte async copy global->LDS per lane; tracked by ASYNCcnt.
__device__ __forceinline__ void async_ld_b128(unsigned lds_byte_off, const void* g) {
  asm volatile("global_load_async_to_lds_b128 %0, %1, off"
               :: "v"(lds_byte_off), "v"((unsigned long long)(size_t)g)
               : "memory");
}
__device__ __forceinline__ void wait_async0() {
  asm volatile("s_wait_asynccnt 0" ::: "memory");
}
#endif

// A fragment (16x32 MxK tile, ISA 7.12.2): lane = row | (hi<<4).
// element e: k = e + (e>=8 ? 8 : 0) + hi*8.  LDS panel row stride = 64B.
__device__ __forceinline__ v16bf lds_a_frag(const uint4* sA, int m, int hi) {
  union { v16bf v; uint4 u[2]; } r;
  r.u[0] = sA[m * 4 + hi];        // bytes m*64 + hi*16
  r.u[1] = sA[m * 4 + hi + 2];    // bytes m*64 + hi*16 + 32
  return r.v;
}
// B fragment (32x16 KxN tile): lane = col | (hi<<4); element e: k = e + (e>=8 ? 8 : 0) + hi*8.
__device__ __forceinline__ v16bf lds_b_frag(const uint4* sB, int n, int hi) {
  union { v16bf v; uint4 u[2]; } r;
  r.u[0] = sB[n * 4 + hi];
  r.u[1] = sB[n * 4 + hi + 2];
  return r.v;
}

struct GemmP {
  const bf16_t* A;   // [M,K] row-major (batched via sA1/sA2)
  const bf16_t* Bt;  // B transposed: [N,K] row-major (batched via sB1/sB2)
  void*         C;   // f32 or bf16 depending on MODE
  const float*  bias;
  int lda, ldb, ldc, M, N, K, bdiv;
  long long sA1, sA2, sB1, sB2, sC1, sC2;
};

// MODE: 0 = f32 store, 1 = bf16 store, 2 = f32 residual accumulate (C += A*B+b),
//       3 = exact-GELU -> bf16, 4 = V^T store vt[b][h][dk][s] (bf16).
// Block = 256 threads (8 waves as 4m x 2n), block tile 128x64; wave tile 32x32.
// All launch shapes divide exactly -> EXEC stays all-ones (WMMA requirement).
template<int MODE>
__global__ __launch_bounds__(256) void wmma_gemm(GemmP p) {
  __shared__ uint4 sA[512];   // 128 rows x 32 k  (8 KB)
  __shared__ uint4 sB[256];   //  64 rows x 32 k  (4 KB)

  const int z  = blockIdx.z;
  const int zq = z / p.bdiv, zr = z % p.bdiv;
  const bf16_t* Ablk = p.A  + zq * p.sA1 + zr * p.sA2
                     + (long long)blockIdx.y * 128 * p.lda;
  const bf16_t* Bblk = p.Bt + zq * p.sB1 + zr * p.sB2
                     + (long long)blockIdx.x * 64 * p.ldb;
  const long long cbase = zq * p.sC1 + zr * p.sC2;

  const int tid  = threadIdx.x;
  const int wave = tid >> 5;
  const int lane = tid & 31;
  const int mw = wave >> 1;          // 0..3 : 32-row slice
  const int nw = wave & 1;           // 0..1 : 32-col slice
  const int r  = lane & 15;
  const int hi = lane >> 4;

  // panel-staging assignments (one b128 per copy)
  const int arow = tid >> 1, aseg = tid & 1;   // A: 128 rows x 2 segs
  const int brow = tid >> 2, bseg = tid & 3;   // B:  64 rows x 4 segs
  const bf16_t* gA = Ablk + (long long)arow * p.lda + aseg * 16;
  const bf16_t* gB = Bblk + (long long)brow * p.ldb + bseg * 8;
#if USE_ASYNC_LDS
  const unsigned sAoff = (unsigned)(size_t)&sA[0];
  const unsigned sBoff = (unsigned)(size_t)&sB[0];
  const unsigned ldsA0 = sAoff + (unsigned)(arow * 4 + aseg * 2) * 16;
  const unsigned ldsB0 = sBoff + (unsigned)(brow * 4 + bseg) * 16;
#endif

  v8f acc[2][2];
#pragma unroll
  for (int i = 0; i < 2; ++i)
#pragma unroll
    for (int j = 0; j < 2; ++j)
#pragma unroll
      for (int v = 0; v < 8; ++v) acc[i][j][v] = 0.0f;

  const int ml0 = mw * 32 + r;
  const int nl0 = nw * 32 + r;

  for (int k = 0; k < p.K; k += 32) {
#if USE_ASYNC_LDS
    async_ld_b128(ldsA0,      gA + k);
    async_ld_b128(ldsA0 + 16, gA + k + 8);
    async_ld_b128(ldsB0,      gB + k);
    if (k + 32 < p.K) {                      // global_prefetch_b8 next panel
      __builtin_prefetch(gA + k + 32, 0, 3);
      __builtin_prefetch(gB + k + 32, 0, 3);
    }
    wait_async0();
#else
    const uint4* g0 = (const uint4*)(gA + k);
    sA[arow * 4 + aseg * 2]     = g0[0];
    sA[arow * 4 + aseg * 2 + 1] = g0[1];
    sB[brow * 4 + bseg] = *(const uint4*)(gB + k);
#endif
    __syncthreads();

    v16bf a0 = lds_a_frag(sA, ml0, hi);
    v16bf a1 = lds_a_frag(sA, ml0 + 16, hi);
    v16bf b0 = lds_b_frag(sB, nl0, hi);
    v16bf b1 = lds_b_frag(sB, nl0 + 16, hi);
    acc[0][0] = __builtin_amdgcn_wmma_f32_16x16x32_bf16(
        false, a0, false, b0, (short)0, acc[0][0], false, false);
    acc[0][1] = __builtin_amdgcn_wmma_f32_16x16x32_bf16(
        false, a0, false, b1, (short)0, acc[0][1], false, false);
    acc[1][0] = __builtin_amdgcn_wmma_f32_16x16x32_bf16(
        false, a1, false, b0, (short)0, acc[1][0], false, false);
    acc[1][1] = __builtin_amdgcn_wmma_f32_16x16x32_bf16(
        false, a1, false, b1, (short)0, acc[1][1], false, false);

    __syncthreads();
  }

  const int mrow0 = blockIdx.y * 128 + mw * 32;
  const int ncol0 = blockIdx.x * 64 + nw * 32;
#pragma unroll
  for (int tj = 0; tj < 2; ++tj) {
    const int n = ncol0 + tj * 16 + r;          // column owned by this lane
    const float bv = p.bias ? p.bias[n] : 0.0f;
#pragma unroll
    for (int ti = 0; ti < 2; ++ti) {
#pragma unroll
      for (int v = 0; v < 8; ++v) {
        const int m = mrow0 + ti * 16 + v + hi * 8;   // lanes16-31 -> M+8
        float val = acc[ti][tj][v] + bv;
        const long long idx = cbase + (long long)m * p.ldc + n;
        if (MODE == 0) {
          ((float*)p.C)[idx] = val;
        } else if (MODE == 1) {
          ((bf16_t*)p.C)[idx] = f2bf(val);
        } else if (MODE == 2) {
          float* Cf = (float*)p.C; Cf[idx] += val;
        } else if (MODE == 3) {
          float g = 0.5f * val * (1.0f + erff(val * 0.70710678118654752f));
          ((bf16_t*)p.C)[idx] = f2bf(g);
        } else { // MODE 4: vt[((b*8+h)*64+dk)*128 + s], m=b*128+s, n=h*64+dk
          const long long vtx = (long long)(m >> 7) * (D_MODEL * SEQ_LEN)
                              + (long long)n * SEQ_LEN + (m & 127);
          ((bf16_t*)p.C)[vtx] = f2bf(val);
        }
      }
    }
  }
}

// fp32 [K,N] -> bf16 transposed [N,K]
__global__ void wconvT_k(const float* __restrict__ W, bf16_t* __restrict__ Wt,
                         int K, int N) {
  int idx = blockIdx.x * 256 + threadIdx.x;
  if (idx >= K * N) return;
  int k = idx / N, n = idx % N;
  Wt[(long long)n * K + k] = f2bf(W[idx]);
}

// 8x8 avg-pool of 4x4 blocks: [32,512,32,32] -> [32,512,8,8]
__global__ void avgpool_k(const float* __restrict__ in, float* __restrict__ out) {
  int idx = blockIdx.x * 256 + threadIdx.x;          // 32*512*64
  int hw = idx & 63, oy = hw >> 3, ox = hw & 7;
  int bc = idx >> 6;
  const float* ip = in + (long long)bc * 1024 + (oy * 4) * 32 + ox * 4;
  float s = 0.f;
#pragma unroll
  for (int dy = 0; dy < 4; ++dy)
#pragma unroll
    for (int dx = 0; dx < 4; ++dx) s += ip[dy * 32 + dx];
  out[idx] = s * 0.0625f;
}

// Closed-form "highpass": the zeroed fftshift block [7:8,7:8] is the single
// frequency bin (3,3).  hp = x - Re(F33 * e^{i th})/64, th=(3pi/4)(r+c).
// Writes hp * x (the conv1 input).
__global__ void highpass_k(const float* __restrict__ p, float* __restrict__ hpm) {
  int ch = blockIdx.x;         // b*512 + c, 16384 channels
  int t  = threadIdx.x;        // 64 threads
  int r = t >> 3, c = t & 7;
  float x  = p[(long long)ch * 64 + t];
  float th = 0.78539816339744831f * (float)((3 * (r + c)) & 7);  // mod 2pi
  float ct = __cosf(th), st = __sinf(th);
  __shared__ float sr[64], si[64];
  sr[t] = x * ct;
  si[t] = -x * st;
  __syncthreads();
  for (int s = 32; s > 0; s >>= 1) {
    if (t < s) { sr[t] += sr[t + s]; si[t] += si[t + s]; }
    __syncthreads();
  }
  float Fr = sr[0], Fi = si[0];
  float hp = x - (Fr * ct - Fi * st) * (1.0f / 64.0f);
  hpm[(long long)ch * 64 + t] = hp * x;
}

// sigmoid(1x1 conv 512->8): out[b,o,hw]
__global__ void conv1_sig_k(const float* __restrict__ in, const float* __restrict__ w1,
                            float* __restrict__ Mout) {
  int idx = blockIdx.x * 256 + threadIdx.x;          // 32*8*64
  if (idx >= 32 * 8 * 64) return;
  int hw = idx & 63, o = (idx >> 6) & 7, b = idx >> 9;
  const float* ip = in + (long long)b * 512 * 64 + hw;
  const float* wp = w1 + o * 512;
  float s = 0.f;
  for (int c = 0; c < 512; ++c) s += ip[c * 64] * wp[c];
  Mout[idx] = 1.0f / (1.0f + __expf(-s));
}

// pairwise-mask loss over l=576 rows of 64: (||sum||^2 - sum||.||^2)/2/(l(l-1))
__global__ void loss_k(const float* rgbM, const float* irM,
                       const float* rgbHM, const float* irHM, float* out) {
  int j = threadIdx.x;                               // 64 threads
  float s = 0.f, sq = 0.f;
  for (int i = 0; i < 256; ++i) { float v = rgbM[i * 64 + j];  s += v; sq += v * v; }
  for (int i = 0; i < 256; ++i) { float v = irM[i * 64 + j];   s += v; sq += v * v; }
  for (int i = 0; i < 32;  ++i) { float v = rgbHM[i * 64 + j]; s += v; sq += v * v; }
  for (int i = 0; i < 32;  ++i) { float v = irHM[i * 64 + j];  s += v; sq += v * v; }
  __shared__ float ss[64], sv[64];
  ss[j] = s * s; sv[j] = sq;
  __syncthreads();
  for (int st = 32; st > 0; st >>= 1) {
    if (j < st) { ss[j] += ss[j + st]; sv[j] += sv[j + st]; }
    __syncthreads();
  }
  if (j == 0) out[0] = (ss[0] - sv[0]) * (0.5f / (576.0f * 575.0f));
}

// x[b, toff+hw, c] = pos[toff+hw, c] + (sum_o M[b,o,hw]*w2[c,o]) * pool[b,c,hw]
__global__ void conv2_tok_k(const float* __restrict__ Mm, const float* __restrict__ pool,
                            const float* __restrict__ w2, const float* __restrict__ pos,
                            float* __restrict__ x, int toff) {
  int idx = blockIdx.x * 256 + threadIdx.x;          // 32*512*64
  int hw = idx & 63, c = (idx >> 6) & 511, b = idx >> 15;
  float s = 0.f;
#pragma unroll
  for (int o = 0; o < 8; ++o) s += Mm[(b * 8 + o) * 64 + hw] * w2[c * 8 + o];
  float val = s * pool[idx];
  int t = toff + hw;
  x[((long long)b * SEQ_LEN + t) * D_MODEL + c] = pos[(long long)t * D_MODEL + c] + val;
}

// One wave32 per 512-wide row; shfl_xor reductions.
template<bool BF16OUT>
__global__ void layernorm_k(const float* __restrict__ x, const float* __restrict__ w,
                            const float* __restrict__ bia, void* out) {
  long long row = blockIdx.x * 8 + (threadIdx.x >> 5);
  int lane = threadIdx.x & 31;
  const float* xr = x + row * D_MODEL;
  float vals[16], s = 0.f, sq = 0.f;
#pragma unroll
  for (int i = 0; i < 16; ++i) {
    float v = xr[lane + i * 32]; vals[i] = v; s += v; sq += v * v;
  }
  for (int m = 16; m >= 1; m >>= 1) {
    s  += __shfl_xor(s, m, 32);
    sq += __shfl_xor(sq, m, 32);
  }
  float mean = s * (1.0f / 512.0f);
  float var  = sq * (1.0f / 512.0f) - mean * mean;
  float rstd = rsqrtf(var + 1e-5f);
#pragma unroll
  for (int i = 0; i < 16; ++i) {
    int j = lane + i * 32;
    float o = (vals[i] - mean) * rstd * w[j] + bia[j];
    if (BF16OUT) ((bf16_t*)out)[row * D_MODEL + j] = f2bf(o);
    else         ((float*)out)[row * D_MODEL + j] = o;
  }
}

// softmax over 128-wide rows of raw scores, scale 1/sqrt(64); bf16 out.
__global__ void softmax_k(const float* __restrict__ att, bf16_t* __restrict__ attb) {
  long long row = blockIdx.x * 8 + (threadIdx.x >> 5); // 32768 rows
  int lane = threadIdx.x & 31;
  const float* ar = att + row * 128;
  float v[4], mx = -1e30f;
#pragma unroll
  for (int i = 0; i < 4; ++i) { v[i] = ar[lane + i * 32] * 0.125f; mx = fmaxf(mx, v[i]); }
  for (int m = 16; m >= 1; m >>= 1) mx = fmaxf(mx, __shfl_xor(mx, m, 32));
  float s = 0.f;
#pragma unroll
  for (int i = 0; i < 4; ++i) { v[i] = __expf(v[i] - mx); s += v[i]; }
  for (int m = 16; m >= 1; m >>= 1) s += __shfl_xor(s, m, 32);
  float inv = 1.0f / s;
#pragma unroll
  for (int i = 0; i < 4; ++i) attb[row * 128 + lane + i * 32] = f2bf(v[i] * inv);
}

// bilinear 8x8 -> 32x32, half-pixel centers (upsampling: no antialias).
__global__ void resize_k(const float* __restrict__ xf, float* __restrict__ out) {
  int idx = blockIdx.x * 256 + threadIdx.x;          // 2*32*512*1024 = 2^25
  int ox = idx & 31, oy = (idx >> 5) & 31;
  int d = (idx >> 10) & 511;
  int b = (idx >> 19) & 31;
  int strm = idx >> 24;
  float fy = (oy + 0.5f) * 0.25f - 0.5f;
  float fx = (ox + 0.5f) * 0.25f - 0.5f;
  int y0 = (int)floorf(fy), x0 = (int)floorf(fx);
  float wy = fy - (float)y0, wx = fx - (float)x0;
  int y0c = max(0, min(7, y0)),     y1c = max(0, min(7, y0 + 1));
  int x0c = max(0, min(7, x0)),     x1c = max(0, min(7, x0 + 1));
  const float* base = xf + ((long long)b * SEQ_LEN + strm * 64) * D_MODEL + d;
  float g00 = base[(y0c * 8 + x0c) * D_MODEL];
  float g01 = base[(y0c * 8 + x1c) * D_MODEL];
  float g10 = base[(y1c * 8 + x0c) * D_MODEL];
  float g11 = base[(y1c * 8 + x1c) * D_MODEL];
  float val = (1.f - wy) * ((1.f - wx) * g00 + wx * g01)
            +         wy * ((1.f - wx) * g10 + wx * g11);
  out[(long long)strm * 16777216 + ((long long)(b * 512 + d) * 1024) + oy * 32 + ox] = val;
}

// ---------------------------------------------------------------------------

extern "C" void kernel_launch(void* const* d_in, const int* in_sizes, int n_in,
                              void* d_out, int out_size, void* d_ws, size_t ws_size,
                              hipStream_t stream) {
  (void)in_sizes; (void)n_in; (void)out_size; (void)ws_size;
  const float* rgb_fea = (const float*)d_in[0];
  const float* ir_fea  = (const float*)d_in[1];
  const float* pos_emb = (const float*)d_in[2];
  const float* conv1_w = (const float*)d_in[3];
  const float* conv2_w = (const float*)d_in[4];
  const float* ln1_w = (const float*)d_in[5];
  const float* ln1_b = (const float*)d_in[6];
  const float* Wq = (const float*)d_in[7];
  const float* bq = (const float*)d_in[8];
  const float* Wk = (const float*)d_in[9];
  const float* bk = (const float*)d_in[10];
  const float* Wv = (const float*)d_in[11];
  const float* bv = (const float*)d_in[12];
  const float* Wo = (const float*)d_in[13];
  const float* bo = (const float*)d_in[14];
  const float* ln2_w = (const float*)d_in[15];
  const float* ln2_b = (const float*)d_in[16];
  const float* W1 = (const float*)d_in[17];
  const float* b1 = (const float*)d_in[18];
  const float* W2 = (const float*)d_in[19];
  const float* b2 = (const float*)d_in[20];
  const float* lnf_w = (const float*)d_in[21];
  const float* lnf_b = (const float*)d_in[22];
  float* out = (float*)d_out;

  // ---- workspace carving (~98 MB total) ----
  size_t off = 0;
  auto carve = [&](size_t bytes) -> void* {
    void* p = (char*)d_ws + off;
    off += (bytes + 255) & ~(size_t)255;
    return p;
  };
  float*  rgb_p   = (float*)carve(32u * 512 * 64 * 4);
  float*  ir_p    = (float*)carve(32u * 512 * 64 * 4);
  float*  rgb_hpm = (float*)carve(32u * 512 * 64 * 4);
  float*  ir_hpm  = (float*)carve(32u * 512 * 64 * 4);
  float*  rgb_M   = (float*)carve(32u * 8 * 64 * 4);
  float*  ir_M    = (float*)carve(32u * 8 * 64 * 4);
  float*  rgb_hpM = (float*)carve(32u * 8 * 64 * 4);
  float*  ir_hpM  = (float*)carve(32u * 8 * 64 * 4);
  float*  x   = (float*)carve(4096u * 512 * 4);
  bf16_t* xb  = (bf16_t*)carve(4096u * 512 * 2);
  bf16_t* qb  = (bf16_t*)carve(4096u * 512 * 2);
  bf16_t* kb  = (bf16_t*)carve(4096u * 512 * 2);
  bf16_t* vt  = (bf16_t*)carve(4096u * 512 * 2);
  float*  att = (float*)carve(256u * 128 * 128 * 4);
  bf16_t* attb= (bf16_t*)carve(256u * 128 * 128 * 2);
  bf16_t* ob  = (bf16_t*)carve(4096u * 512 * 2);
  bf16_t* hb  = (bf16_t*)carve(4096u * 2048 * 2);
  float*  xf  = (float*)carve(4096u * 512 * 4);
  bf16_t* wq_t = (bf16_t*)carve(512u * 512 * 2);
  bf16_t* wk_t = (bf16_t*)carve(512u * 512 * 2);
  bf16_t* wv_t = (bf16_t*)carve(512u * 512 * 2);
  bf16_t* wo_t = (bf16_t*)carve(512u * 512 * 2);
  bf16_t* w1_t = (bf16_t*)carve(512u * 2048 * 2);
  bf16_t* w2_t = (bf16_t*)carve(2048u * 512 * 2);

  auto gemm = [&](int mode, const bf16_t* A, int lda, long long sA1, long long sA2,
                  const bf16_t* Bt, int ldb, long long sB1, long long sB2,
                  void* C, int ldc, long long sC1, long long sC2,
                  const float* bias, int M, int N, int K, int bdiv, int batch) {
    GemmP p;
    p.A = A; p.Bt = Bt; p.C = C; p.bias = bias;
    p.lda = lda; p.ldb = ldb; p.ldc = ldc;
    p.M = M; p.N = N; p.K = K; p.bdiv = bdiv;
    p.sA1 = sA1; p.sA2 = sA2; p.sB1 = sB1; p.sB2 = sB2; p.sC1 = sC1; p.sC2 = sC2;
    dim3 grid(N / 64, M / 128, batch), blk(256, 1, 1);
    switch (mode) {
      case 0: wmma_gemm<0><<<grid, blk, 0, stream>>>(p); break;
      case 1: wmma_gemm<1><<<grid, blk, 0, stream>>>(p); break;
      case 2: wmma_gemm<2><<<grid, blk, 0, stream>>>(p); break;
      case 3: wmma_gemm<3><<<grid, blk, 0, stream>>>(p); break;
      default: wmma_gemm<4><<<grid, blk, 0, stream>>>(p); break;
    }
  };

  // ---- frequency-guided mask path ----
  avgpool_k<<<4096, 256, 0, stream>>>(rgb_fea, rgb_p);
  avgpool_k<<<4096, 256, 0, stream>>>(ir_fea, ir_p);
  highpass_k<<<16384, 64, 0, stream>>>(rgb_p, rgb_hpm);
  highpass_k<<<16384, 64, 0, stream>>>(ir_p, ir_hpm);
  conv1_sig_k<<<64, 256, 0, stream>>>(rgb_hpm, conv1_w, rgb_hpM);
  conv1_sig_k<<<64, 256, 0, stream>>>(ir_hpm, conv1_w, ir_hpM);
  conv1_sig_k<<<64, 256, 0, stream>>>(rgb_p, conv1_w, rgb_M);
  conv1_sig_k<<<64, 256, 0, stream>>>(ir_p, conv1_w, ir_M);
  conv2_tok_k<<<4096, 256, 0, stream>>>(rgb_M, rgb_p, conv2_w, pos_emb, x, 0);
  conv2_tok_k<<<4096, 256, 0, stream>>>(ir_M, ir_p, conv2_w, pos_emb, x, 64);
  loss_k<<<1, 64, 0, stream>>>(rgb_M, ir_M, rgb_hpM, ir_hpM, out + 33554432);

  // ---- 8 transformer layers, all GEMMs on v_wmma_f32_16x16x32_bf16 ----
  for (int l = 0; l < NLAYER; ++l) {
    wconvT_k<<<1024, 256, 0, stream>>>(Wq + (long long)l * 262144, wq_t, 512, 512);
    wconvT_k<<<1024, 256, 0, stream>>>(Wk + (long long)l * 262144, wk_t, 512, 512);
    wconvT_k<<<1024, 256, 0, stream>>>(Wv + (long long)l * 262144, wv_t, 512, 512);
    wconvT_k<<<1024, 256, 0, stream>>>(Wo + (long long)l * 262144, wo_t, 512, 512);
    wconvT_k<<<4096, 256, 0, stream>>>(W1 + (long long)l * 1048576, w1_t, 512, 2048);
    wconvT_k<<<4096, 256, 0, stream>>>(W2 + (long long)l * 1048576, w2_t, 2048, 512);

    layernorm_k<true><<<512, 256, 0, stream>>>(x, ln1_w + l * 512, ln1_b + l * 512, xb);
    gemm(1, xb, 512, 0, 0, wq_t, 512, 0, 0, qb, 512, 0, 0, bq + l * 512, 4096, 512, 512, 1, 1);
    gemm(1, xb, 512, 0, 0, wk_t, 512, 0, 0, kb, 512, 0, 0, bk + l * 512, 4096, 512, 512, 1, 1);
    gemm(4, xb, 512, 0, 0, wv_t, 512, 0, 0, vt, 512, 0, 0, bv + l * 512, 4096, 512, 512, 1, 1);
    // scores[b,h] = Q[128,64] * K^T  (Bt = K rows, batched over 256 (b,h))
    gemm(0, qb, 512, 65536, 64, kb, 512, 65536, 64,
         att, 128, 131072, 16384, nullptr, 128, 128, 64, 8, 256);
    softmax_k<<<4096, 256, 0, stream>>>(att, attb);
    // O[b,h] = P[128,128] * V ; Bt = V^T[64,128] written by MODE 4
    gemm(1, attb, 128, 131072, 16384, vt, 128, 65536, 8192,
         ob, 512, 65536, 64, nullptr, 128, 64, 128, 8, 256);
    gemm(2, ob, 512, 0, 0, wo_t, 512, 0, 0, x, 512, 0, 0, bo + l * 512, 4096, 512, 512, 1, 1);

    layernorm_k<true><<<512, 256, 0, stream>>>(x, ln2_w + l * 512, ln2_b + l * 512, xb);
    gemm(3, xb, 512, 0, 0, w1_t, 512, 0, 0, hb, 2048, 0, 0, b1 + l * 2048, 4096, 2048, 512, 1, 1);
    gemm(2, hb, 2048, 0, 0, w2_t, 2048, 0, 0, x, 512, 0, 0, b2 + l * 512, 4096, 512, 2048, 1, 1);
  }

  layernorm_k<false><<<512, 256, 0, stream>>>(x, lnf_w, lnf_b, xf);
  resize_k<<<131072, 256, 0, stream>>>(xf, out);
}
